// LatentScaleSelectionHead_28518582845720
// MI455X (gfx1250) — matrix operand order repaired and checked by reading, not compiled, not run on hardware
//
#include <hip/hip_runtime.h>

typedef __attribute__((ext_vector_type(16))) _Float16 v16h;
typedef __attribute__((ext_vector_type(8)))  float    v8f;

#define N_DIM   1325
#define NN      (1325*1325)          // 1755625
#define NPAD    1344                 // 42 * 32
#define K_CLEAN 1312                 // 41 * 32 ; all k < 1312 are < N_DIM
#define GSTRIDE 1344
#define NBINS   2048
#define KEEP_CNT 526688              // NN - int(NN*0.7)
#define NUM_PTS 300
#define CAM_BASE 0                   // multiple_cams: 32*4*1024 floats
#define LBL_BASE 131072              // gt_labels: 32
#define PTS_BASE 131104              // gt_points: 64
#define BOX_BASE 131168              // gt_bboxes_: 128

// ---------------- workspace zero (histograms only) ----------------
__global__ void zero_hist_kernel(int* __restrict__ hist) {
    int i = blockIdx.x * blockDim.x + threadIdx.x;
    if (i < 8 * NBINS) hist[i] = 0;
}

// ---------------- per-map histogram (LDS privatized) ----------------
__global__ __launch_bounds__(256) void hist_kernel(const float* __restrict__ attns,
                                                   int* __restrict__ hist) {
    __shared__ int sh[NBINS];
    for (int j = threadIdx.x; j < NBINS; j += blockDim.x) sh[j] = 0;
    __syncthreads();

    int m   = blockIdx.x;                  // map id 0..7
    int blk = 8 + (m >> 1), b = m & 1;
    const float* A = attns + (size_t)(blk * 2 + b) * (size_t)NN;

    int tid    = blockIdx.y * blockDim.x + threadIdx.x;
    int stride = gridDim.y * blockDim.x;
    for (int i = tid; i < NN; i += stride) {
        float v  = A[i];
        int bin  = (int)(v * (float)NBINS);
        bin      = bin < 0 ? 0 : (bin > NBINS - 1 ? NBINS - 1 : bin);
        atomicAdd(&sh[bin], 1);
    }
    __syncthreads();
    int* h = hist + m * NBINS;
    for (int j = threadIdx.x; j < NBINS; j += blockDim.x) {
        int c = sh[j];
        if (c) atomicAdd(&h[j], c);
    }
}

// ---------------- threshold: cumulative from top ----------------
__global__ void thresh_kernel(const int* __restrict__ hist, float* __restrict__ thr) {
    int m = threadIdx.x;
    if (m >= 8) return;
    const int* h = hist + m * NBINS;
    int acc = 0, bin = NBINS - 1;
    for (; bin >= 0; --bin) {
        acc += h[bin];
        if (acc >= KEEP_CNT) break;
    }
    if (bin < 0) bin = 0;
    thr[m] = (float)bin / (float)NBINS;   // lower edge of selected bin
}

// ---------------- reciprocal row sums of filtered map (+1 for the eye) ----------------
__global__ __launch_bounds__(256) void rowsum_kernel(const float* __restrict__ attns,
                                                     const float* __restrict__ thr,
                                                     float* __restrict__ rinv) {
    int m   = blockIdx.x;
    int blk = 8 + (m >> 1), b = m & 1;
    const float* A = attns + (size_t)(blk * 2 + b) * (size_t)NN;
    float t = thr[m];

    int wave = threadIdx.x >> 5, lane = threadIdx.x & 31;
    int r = blockIdx.y * 8 + wave;
    if (r >= N_DIM) return;
    const float* row = A + (size_t)r * N_DIM;
    float s = 0.f;
    for (int c = lane; c < N_DIM; c += 32) {
        float a = row[c];
        s += (a >= t) ? a : 0.f;
    }
    #pragma unroll
    for (int off = 16; off > 0; off >>= 1) s += __shfl_xor(s, off, 32);
    if (lane == 0) rinv[m * N_DIM + r] = 1.0f / (1.0f + s);   // store RECIPROCAL
}

// ---------------- gather normalized aug11 rows -> G0, emit scale s=3 ----------------
__global__ __launch_bounds__(256) void gatherA_kernel(const float* __restrict__ attns,
                                                      const int* __restrict__ pos_inds,
                                                      const float* __restrict__ thr,
                                                      const float* __restrict__ rinv,
                                                      float* __restrict__ G0,
                                                      float* __restrict__ out) {
    int bi = blockIdx.y;                // 0..31
    int b  = bi >> 4, i = bi & 15;
    int c  = blockIdx.x * blockDim.x + threadIdx.x;
    if (c >= GSTRIDE) return;

    int m = 6 + b;                      // blk 11 -> map id (11-8)*2+b
    const float* A = attns + (size_t)(11 * 2 + b) * (size_t)NN;
    int row = (N_DIM - NUM_PTS) + pos_inds[b * 16 + i];
    float r = rinv[m * N_DIM + row];
    float t = thr[m];

    float val = 0.f;
    if (c < N_DIM) {
        float a = A[(size_t)row * N_DIM + c];
        val = (a >= t) ? a : 0.f;
        if (c == row) val += 1.f;
        val *= r;
    }
    G0[((size_t)b * 16 + i) * GSTRIDE + c] = val;
    if (c >= 1 && c <= 1024)
        out[CAM_BASE + (((size_t)b * 16 + i) * 4 + 3) * 1024 + (c - 1)] = val;
}

// ---------------- one chain stage: Gdst = Gsrc (16xN) @ aug_blk (NxN), emit scale s ----------------
__global__ __launch_bounds__(32) void wmma_stage_kernel(const float* __restrict__ attns,
                                                        const float* __restrict__ thr,
                                                        const float* __restrict__ rinv,
                                                        const float* __restrict__ Gsrc,
                                                        float* __restrict__ Gdst,
                                                        float* __restrict__ out,
                                                        int blk, int s) {
    const int tileN = blockIdx.x;       // 0..82 -> output cols [tileN*16, +16)
    const int b     = blockIdx.y;       // image
    const int lane  = threadIdx.x;      // 0..31
    const int m     = (blk - 8) * 2 + b;

    const float* A  = attns + (size_t)(blk * 2 + b) * (size_t)NN;
    const float* rv = rinv + m * N_DIM;
    const float  t  = thr[m];

    const bool lo = lane < 16;
    const int  nl = lane & 15;
    const int  col  = tileN * 16 + nl;                // this lane's B/D column
    const int  colc = col < N_DIM ? col : N_DIM - 1;  // address-safe clamp only

    const float* gs = Gsrc + ((size_t)b * 16 + nl) * GSTRIDE;  // A row M = nl

    // ---- operand tile builders (documented wave32 f16 WMMA layouts) ----
    auto make_a = [&](int k0, bool masked) -> v16h {
        v16h av;
        #pragma unroll
        for (int v = 0; v < 8; ++v) {
            int kk = ((v < 4) ? 0 : 16) + (lo ? 0 : 8) + 2 * (v & 3);
            int k  = k0 + kk;
            float f0 = gs[k];        // GSTRIDE=1344 covers all padded k
            float f1 = gs[k + 1];
            if (masked) {
                f0 *= (k     < N_DIM) ? 1.f : 0.f;
                f1 *= (k + 1 < N_DIM) ? 1.f : 0.f;
            }
            av[2 * v]     = (_Float16)f0;
            av[2 * v + 1] = (_Float16)f1;
        }
        return av;
    };
    auto make_b = [&](int k0, bool masked) -> v16h {
        v16h bv;
        #pragma unroll
        for (int v = 0; v < 8; ++v) {
            int k = k0 + ((v < 4) ? 0 : 16) + (lo ? 0 : 8) + 2 * (v & 3);
            #pragma unroll
            for (int q = 0; q < 2; ++q) {
                int kq = k + q;
                int kc = masked ? (kq < N_DIM ? kq : N_DIM - 1) : kq;
                float a = A[(size_t)kc * N_DIM + colc];
                float f = (a >= t) ? a : 0.f;
                f += (kq == col) ? 1.f : 0.f;          // + eye
                f *= rv[kc];                           // row-normalize (reciprocal mul)
                if (masked) f *= (kq < N_DIM) ? 1.f : 0.f;
                bv[2 * v + q] = (_Float16)f;
            }
        }
        return bv;
    };

    v8f acc0 = {}, acc1 = {};
    int k0 = 0;
    // main: pairs of clean iterations on independent accumulators (hide WMMA RAW hazard)
    for (; k0 + 64 <= K_CLEAN; k0 += 64) {
        acc0 = __builtin_amdgcn_wmma_f32_16x16x32_f16(false, make_a(k0, false),      false, make_b(k0, false),
                                                      (short)0, acc0, false, false);
        acc1 = __builtin_amdgcn_wmma_f32_16x16x32_f16(false, make_a(k0 + 32, false), false, make_b(k0 + 32, false),
                                                      (short)0, acc1, false, false);
    }
    // remaining clean iteration (K_CLEAN = 41*32, odd count)
    for (; k0 + 32 <= K_CLEAN; k0 += 32) {
        acc0 = __builtin_amdgcn_wmma_f32_16x16x32_f16(false, make_a(k0, false), false, make_b(k0, false),
                                                      (short)0, acc0, false, false);
    }
    // masked tail: k in [1312, 1344), zero k >= N_DIM
    acc1 = __builtin_amdgcn_wmma_f32_16x16x32_f16(false, make_a(K_CLEAN, true), false, make_b(K_CLEAN, true),
                                                  (short)0, acc1, false, false);
    v8f acc = acc0 + acc1;

    // ---- store D: VGPR v -> (M = v + 8*hi, N = nl)
    #pragma unroll
    for (int v = 0; v < 8; ++v) {
        int M = v + (lo ? 0 : 8);
        Gdst[((size_t)b * 16 + M) * GSTRIDE + col] = acc[v];
        if (col >= 1 && col <= 1024)
            out[CAM_BASE + (((size_t)b * 16 + M) * 4 + s) * 1024 + (col - 1)] = acc[v];
    }
}

// ---------------- tail: gt_labels / gt_points / gt_bboxes ----------------
__global__ void tail_kernel(const int* __restrict__ pos_inds,
                            const int* __restrict__ pos_gt,
                            const int* __restrict__ labels,
                            const float* __restrict__ points,
                            const float* __restrict__ gt_bboxes,
                            float* __restrict__ out) {
    int tid = threadIdx.x;
    if (tid < 32) {                               // labels
        int b = tid >> 4, i = tid & 15;
        int p = pos_inds[b * 16 + i];
        out[LBL_BASE + tid] = (float)labels[b * NUM_PTS + p];
    }
    if (tid < 64) {                               // points
        int b = tid >> 5, r = (tid >> 1) & 15, d = tid & 1;
        int p = pos_inds[b * 16 + r];
        out[PTS_BASE + tid] = points[((size_t)b * NUM_PTS + p) * 2 + d];
    }
    if (tid < 128) {                              // bboxes
        int b = tid >> 6, r = (tid >> 2) & 15, d = tid & 3;
        int g = pos_gt[b * 16 + r];
        out[BOX_BASE + tid] = gt_bboxes[((size_t)b * 16 + g) * 4 + d];
    }
}

extern "C" void kernel_launch(void* const* d_in, const int* in_sizes, int n_in,
                              void* d_out, int out_size, void* d_ws, size_t ws_size,
                              hipStream_t stream) {
    (void)in_sizes; (void)n_in; (void)out_size; (void)ws_size;
    const float* attns     = (const float*)d_in[0];
    const int*   pos_inds  = (const int*)  d_in[1];
    const int*   pos_gt    = (const int*)  d_in[2];
    const int*   labels    = (const int*)  d_in[3];
    const float* points    = (const float*)d_in[4];
    const float* gt_bboxes = (const float*)d_in[5];
    float* out = (float*)d_out;

    char* w = (char*)d_ws;
    int*   hist = (int*)  (w + 0);            // 8*2048*4   = 65536 B
    float* thr  = (float*)(w + 65536);        // 8*4        (pad to 65568)
    float* rinv = (float*)(w + 65568);        // 8*1325*4   = 42400 B -> 107968
    float* G0   = (float*)(w + 108032);       // 2*16*1344*4 = 172032 B
    float* G1   = (float*)(w + 108032 + 172032);

    zero_hist_kernel<<<dim3((8 * NBINS + 255) / 256), 256, 0, stream>>>(hist);
    hist_kernel    <<<dim3(8, 128), 256, 0, stream>>>(attns, hist);
    thresh_kernel  <<<1, 32, 0, stream>>>(hist, thr);
    rowsum_kernel  <<<dim3(8, (N_DIM + 7) / 8), 256, 0, stream>>>(attns, thr, rinv);
    gatherA_kernel <<<dim3((GSTRIDE + 255) / 256, 32), 256, 0, stream>>>(attns, pos_inds, thr, rinv, G0, out);

    // chain: s=3 is aug11 itself (gather); then multiply by aug10, aug9, aug8
    wmma_stage_kernel<<<dim3(83, 2), 32, 0, stream>>>(attns, thr, rinv, G0, G1, out, 10, 2);
    wmma_stage_kernel<<<dim3(83, 2), 32, 0, stream>>>(attns, thr, rinv, G1, G0, out,  9, 1);
    wmma_stage_kernel<<<dim3(83, 2), 32, 0, stream>>>(attns, thr, rinv, G0, G1, out,  8, 0);

    tail_kernel<<<1, 256, 0, stream>>>(pos_inds, pos_gt, labels, points, gt_bboxes, out);
}
